// GRU_4904852652925
// MI455X (gfx1250) — matrix operand order repaired and checked by reading, not compiled, not run on hardware
//
#include <hip/hip_runtime.h>
#include <hip/hip_bf16.h>

typedef __attribute__((ext_vector_type(16))) __bf16 v16bf;
typedef __attribute__((ext_vector_type(8)))  __bf16 v8bf;
typedef __attribute__((ext_vector_type(8)))  float  v8f;
typedef __attribute__((ext_vector_type(4)))  unsigned int u32x4;
typedef __attribute__((ext_vector_type(8)))  int      i32x8;
typedef __attribute__((ext_vector_type(4)))  int      i32x4;

namespace {
constexpr int B_ = 512;   // batch
constexpr int T_ = 512;   // time steps
constexpr int D_ = 128;   // input dim
constexpr int H_ = 100;   // hidden dim
constexpr int NT = 7;     // WMMA column tiles (N padded 100 -> 112)
constexpr int KP = 128;   // K padded (h-part 100->128; x-part exactly 128)
constexpr int WROW = 228; // H + D
constexpr int NBLK = B_ / 16;            // 32 batch blocks
constexpr int CHUNK = 3 * NT * 32 * 8;   // 5376 f32 = gate pre-activations per (t, blk)
}

#if defined(__has_builtin)
#if __has_builtin(__builtin_amdgcn_tensor_load_to_lds) && \
    __has_builtin(__builtin_amdgcn_s_wait_tensorcnt)
#define HAVE_TDM 1
#endif
#endif

// ---------------------------------------------------------------- WMMA helpers

__device__ __forceinline__ v8f wmma_bf16(v16bf a, v16bf b, v8f c) {
  return __builtin_amdgcn_wmma_f32_16x16x32_bf16(
      /*neg_a=*/false, a, /*neg_b=*/false, b,
      /*c_mod=*/(short)0, c, /*reuse_a=*/false, /*reuse_b=*/false);
}

// 16x32 bf16 A-fragment from row-major [16][stride]: lanes 0-15 row=lane,
// K chunks {0..7,16..23}; lanes 16-31 row=lane-16, K chunks {8..15,24..31}.
__device__ __forceinline__ v16bf load_a_frag(const __bf16* base, int stride, int kstep) {
  const int lane = threadIdx.x & 31;
  const int row  = lane & 15;
  const int hi   = lane >> 4;
  const __bf16* p = base + row * stride + kstep * 32 + hi * 8;
  v8bf c0 = *(const v8bf*)(p);
  v8bf c1 = *(const v8bf*)(p + 16);
  return __builtin_shufflevector(c0, c1, 0,1,2,3,4,5,6,7,8,9,10,11,12,13,14,15);
}

// Pre-packed B fragments: one contiguous 32B load per lane.
__device__ __forceinline__ v16bf load_b_frag(const __bf16* wf, int g, int ks, int n) {
  const int lane = threadIdx.x & 31;
  return *(const v16bf*)(wf + ((((g * 4 + ks) * NT) + n) << 9) + (lane << 4));
}

__device__ __forceinline__ float sigmoidf_(float v) { return 1.f / (1.f + __expf(-v)); }
__device__ __forceinline__ float tanhf_(float v) {
  float e = __expf(2.f * v);
  return 1.f - 2.f / (e + 1.f);
}

// ---------------------------------------------------------------- TDM helper

#ifdef HAVE_TDM
// 1-D TDM copy of `CHUNK` f32 from global to LDS, per D# bitfields in
// CDNA5 ISA (group0: count/lds/global/type; group1:
// data_size=4B, tensor_dim0=tile_dim0=CHUNK, 1-D tile).
// This toolchain declares the 6-arg (clang-23 / therock-10.0) form:
//   (u32x4 g0, i32x8 g1, i32x4, i32x4, i32x8, i32 cpol)
__device__ __forceinline__ void tdm_load_chunk(unsigned int lds_off, const float* gsrc) {
  const unsigned long long ga = (unsigned long long)(uintptr_t)gsrc;
  u32x4 g0;
  g0[0] = 1u;                                            // count=1 (valid user D#)
  g0[1] = lds_off;                                       // lds_addr (bytes)
  g0[2] = (unsigned int)ga;                              // global_addr[31:0]
  g0[3] = (unsigned int)((ga >> 32) & 0x1FFFFFFull) | (2u << 30);  // addr[56:32] | type=2
  i32x8 g1;
  g1[0] = (int)(2u << 16);                               // data_size = 4 bytes
  g1[1] = (int)((unsigned)(CHUNK & 0xFFFF) << 16);       // tensor_dim0[15:0]
  g1[2] = (int)(1u << 16);                               // tensor_dim0[31:16]=0 | tensor_dim1=1
  g1[3] = (int)((unsigned)CHUNK << 16);                  // tile_dim0 = CHUNK
  g1[4] = 0;                                             // tile_dim1/2 unused
  g1[5] = (int)CHUNK;                                    // tensor_dim0_stride
  g1[6] = 0;
  g1[7] = 0;
  i32x4 gz4 = {0, 0, 0, 0};                              // groups 2/3 unused (<=2D)
  i32x8 gz8 = {0, 0, 0, 0, 0, 0, 0, 0};                  // trailing group (zero-filled)
  __builtin_amdgcn_tensor_load_to_lds(g0, g1, gz4, gz4, gz8, 0);
}
#endif

// ---------------------------------------------------------------- prep kernels

// f32 [B,T,D] -> bf16 [T,B,D] (transpose during conversion; both sides coalesced)
__global__ void k_convert_xT(const float* __restrict__ x, __bf16* __restrict__ xbf, int n) {
  for (int o = blockIdx.x * blockDim.x + threadIdx.x; o < n; o += gridDim.x * blockDim.x) {
    const int t = o / (B_ * D_);
    const int r = o % (B_ * D_);
    const int b = r / D_;
    const int d = r % D_;
    xbf[o] = (__bf16)x[(b * T_ + t) * D_ + d];
  }
}

// Pack Wz/Wr/Wh into per-lane WMMA B-fragment order (bf16, zero-padded).
__global__ void k_pack_w(const float* __restrict__ Wz, const float* __restrict__ Wr,
                         const float* __restrict__ Wh,
                         __bf16* __restrict__ whf, __bf16* __restrict__ wxf) {
  const int per_part = 3 * 4 * NT * 512;   // 43008
  const int total = 2 * per_part;
  for (int idx = blockIdx.x * blockDim.x + threadIdx.x; idx < total;
       idx += gridDim.x * blockDim.x) {
    int r = idx;
    const int part = r / per_part;  r %= per_part;
    const int g    = r / 14336;     r %= 14336;
    const int ks   = r / 3584;      r %= 3584;
    const int n    = r / 512;       r %= 512;
    const int lane = r / 16;
    const int i    = r % 16;
    const int k_local = (i < 8) ? ((lane >> 4) * 8 + i) : (16 + (lane >> 4) * 8 + (i - 8));
    const int K   = ks * 32 + k_local;
    const int col = n * 16 + (lane & 15);
    const float* W = (g == 0) ? Wz : (g == 1) ? Wr : Wh;
    float v = 0.f;
    if (part == 0) {
      if (K < H_ && col < H_) v = W[col * WROW + K];
    } else {
      if (col < H_)           v = W[col * WROW + H_ + K];
    }
    const int dst = (((g * 4 + ks) * NT) + n) * 512 + lane * 16 + i;
    (part ? wxf : whf)[dst] = (__bf16)v;
  }
}

// ---------------- phase 1: parallel x-projection, fragment-major f32 output
// XF[t][blk][g][n][lane][8]: one contiguous 21.5KB chunk per (t, batch-block).
__global__ void k_xproj(const __bf16* __restrict__ xbf, const __bf16* __restrict__ wxf,
                        float* __restrict__ XF) {
  const int n    = threadIdx.x >> 5;
  const int lane = threadIdx.x & 31;
  const int t    = blockIdx.x >> 5;        // NBLK = 32
  const int blk  = blockIdx.x & 31;
  const __bf16* abase = xbf + ((size_t)t * B_ + blk * 16) * D_;  // 16x128 contiguous
  v8f az = {}, ar = {}, ah = {};
#pragma unroll
  for (int ks = 0; ks < 4; ++ks) {
    v16bf a = load_a_frag(abase, D_, ks);
    az = wmma_bf16(a, load_b_frag(wxf, 0, ks, n), az);
    ar = wmma_bf16(a, load_b_frag(wxf, 1, ks, n), ar);
    ah = wmma_bf16(a, load_b_frag(wxf, 2, ks, n), ah);
  }
  float* chunk = XF + (size_t)blockIdx.x * CHUNK;
  *(v8f*)(chunk + ((0 * NT + n) << 8) + (lane << 3)) = az;   // 32B/lane, coalesced
  *(v8f*)(chunk + ((1 * NT + n) << 8) + (lane << 3)) = ar;
  *(v8f*)(chunk + ((2 * NT + n) << 8) + (lane << 3)) = ah;
}

// ---------------- phase 2: recurrent scan (32 independent blocks over batch)
__global__ void k_scan(const __bf16* __restrict__ whf, const float* __restrict__ XF,
                       float* __restrict__ out) {
  __shared__ __align__(16) __bf16 hbf[16][KP];
  __shared__ __align__(16) __bf16 rhbf[16][KP];
  __shared__ float hf[16][112];
#ifdef HAVE_TDM
  __shared__ __align__(32) float xstage[2][CHUNK];   // TDM double buffer (43KB)
#endif

  const int n    = threadIdx.x >> 5;
  const int lane = threadIdx.x & 31;
  const int col  = n * 16 + (lane & 15);
  const int rhi  = (lane >> 4) * 8;
  const int blk  = blockIdx.x;
  const int b0   = blk * 16;

  for (int i = threadIdx.x; i < 16 * KP; i += blockDim.x) {
    hbf[i / KP][i % KP]  = (__bf16)0.f;
    rhbf[i / KP][i % KP] = (__bf16)0.f;
  }
  for (int i = threadIdx.x; i < 16 * 112; i += blockDim.x) hf[i / 112][i % 112] = 0.f;

  v16bf bz[4], br[4], bh[4];
#pragma unroll
  for (int ks = 0; ks < 4; ++ks) {
    bz[ks] = load_b_frag(whf, 0, ks, n);
    br[ks] = load_b_frag(whf, 1, ks, n);
    bh[ks] = load_b_frag(whf, 2, ks, n);
  }

#ifdef HAVE_TDM
  if (threadIdx.x < 32) {                       // wave 0 drives the TDM
    tdm_load_chunk((unsigned int)(uintptr_t)&xstage[0][0], XF + (size_t)blk * CHUNK);
    __builtin_amdgcn_s_wait_tensorcnt(0);
  }
#endif
  __syncthreads();

  for (int t = 0; t < T_; ++t) {
#ifdef HAVE_TDM
    if (threadIdx.x < 32 && t + 1 < T_) {       // async-fetch step t+1 during compute
      tdm_load_chunk((unsigned int)(uintptr_t)&xstage[(t + 1) & 1][0],
                     XF + ((size_t)(t + 1) * NBLK + blk) * CHUNK);
    }
    const float* xc = &xstage[t & 1][0];
#else
    const float* xc = XF + ((size_t)t * NBLK + blk) * CHUNK;
    if (t + 1 < T_)                              // global_prefetch_b8 for step t+1
      __builtin_prefetch(XF + ((size_t)(t + 1) * NBLK + blk) * CHUNK + threadIdx.x * 24, 0, 1);
#endif
    const v8f xz = *(const v8f*)(xc + ((0 * NT + n) << 8) + (lane << 3));
    const v8f xr = *(const v8f*)(xc + ((1 * NT + n) << 8) + (lane << 3));
    const v8f xh = *(const v8f*)(xc + ((2 * NT + n) << 8) + (lane << 3));

    // gates z, r:  acc = h @ Wg_h^T
    v8f za = {}, ra = {};
#pragma unroll
    for (int ks = 0; ks < 4; ++ks) {
      v16bf a = load_a_frag(&hbf[0][0], KP, ks);
      za = wmma_bf16(a, bz[ks], za);
      ra = wmma_bf16(a, br[ks], ra);
    }
    float zreg[8];
#pragma unroll
    for (int i = 0; i < 8; ++i) {
      const int row = rhi + i;
      const float z = sigmoidf_(za[i] + xz[i]);
      const float r = sigmoidf_(ra[i] + xr[i]);
      zreg[i] = z;
      rhbf[row][col] = (col < H_) ? (__bf16)(r * hf[row][col]) : (__bf16)0.f;
    }
    __syncthreads();                             // publish r*h

    // candidate: acc = (r*h) @ Wh_h^T
    v8f ha = {};
#pragma unroll
    for (int ks = 0; ks < 4; ++ks) {
      v16bf a = load_a_frag(&rhbf[0][0], KP, ks);
      ha = wmma_bf16(a, bh[ks], ha);
    }
#pragma unroll
    for (int i = 0; i < 8; ++i) {
      const int row = rhi + i;
      const float hh   = tanhf_(ha[i] + xh[i]);
      const float hold = hf[row][col];
      const float z    = zreg[i];
      const float hnew = (1.f - z) * hold + z * hh;
      if (col < H_) {
        out[((size_t)t * B_ + (b0 + row)) * H_ + col] = hnew;
        hf[row][col]  = hnew;
        hbf[row][col] = (__bf16)hnew;
      }
    }
#ifdef HAVE_TDM
    if (threadIdx.x < 32) __builtin_amdgcn_s_wait_tensorcnt(0);  // t+1 chunk landed
#endif
    __syncthreads();                             // publish h (and staged chunk)
  }
}

// ---------------------------------------------------------------- launch

extern "C" void kernel_launch(void* const* d_in, const int* in_sizes, int n_in,
                              void* d_out, int out_size, void* d_ws, size_t ws_size,
                              hipStream_t stream) {
  (void)in_sizes; (void)n_in; (void)out_size; (void)ws_size;
  const float* x  = (const float*)d_in[0];
  const float* Wz = (const float*)d_in[1];
  const float* Wr = (const float*)d_in[2];
  const float* Wh = (const float*)d_in[3];
  float* out = (float*)d_out;

  char* ws = (char*)d_ws;
  __bf16* xbf = (__bf16*)ws;                               // 67,108,864 B
  size_t off = (size_t)B_ * T_ * D_ * sizeof(__bf16);
  __bf16* whf = (__bf16*)(ws + off); off += 86016;
  __bf16* wxf = (__bf16*)(ws + off); off += 86016;
  float* XF = (float*)(ws + off);                          // T*32*CHUNK f32 = 352MB

  const int nx = B_ * T_ * D_;
  k_convert_xT<<<4096, 256, 0, stream>>>(x, xbf, nx);
  k_pack_w<<<336, 256, 0, stream>>>(Wz, Wr, Wh, whf, wxf);
  k_xproj<<<T_ * NBLK, NT * 32, 0, stream>>>(xbf, wxf, XF);
  k_scan<<<NBLK, NT * 32, 0, stream>>>(whf, XF, out);
}
